// MamTrans_56049323213047
// MI455X (gfx1250) — matrix operand order repaired and checked by reading, not compiled, not run on hardware
//
#include <hip/hip_runtime.h>
#include <hip/hip_bf16.h>
#include <math.h>

// ---------------------------------------------------------------------------
// MI455X (gfx1250) implementation. wave32, WMMA 16x16x32 f16 -> f32.
// All matmuls go through v_wmma_f32_16x16x32_f16; epilogues fuse
// bias/activation/residual. LDS reductions for BN/LN/softmax.
// ---------------------------------------------------------------------------

typedef __attribute__((ext_vector_type(16))) _Float16 v16h;
typedef __attribute__((ext_vector_type(8)))  float    v8f;

#define ACT_NONE     0
#define ACT_RELU     1
#define ACT_GELU     2
#define ACT_SILU     3
#define ACT_SOFTPLUS 4

__device__ __forceinline__ float act_apply(float v, int act) {
  switch (act) {
    case ACT_RELU: return fmaxf(v, 0.f);
    case ACT_GELU: {
      float c = v + 0.044715f * v * v * v;
      return 0.5f * v * (1.f + tanhf(0.7978845608028654f * c));
    }
    case ACT_SILU: return v / (1.f + __expf(-v));
    case ACT_SOFTPLUS: return (v > 20.f) ? v : log1pf(__expf(v));
    default: return v;
  }
}

__device__ __forceinline__ float silu_f(float v) { return v / (1.f + __expf(-v)); }

// ---------------------------------------------------------------------------
// Generic batched GEMM via WMMA.  C[z] = act(A[z]*B[z] + bias) + res[z]
// z = z1*bz2 + z2 with independent strides per level (covers per-(b,h)
// attention slices).  TA: A(m,k)=A[k*lda+m].  TB: B(k,n)=B[n*ldb+k].
// Block = 128 threads = 4 waves, each wave owns one 16x16 D tile (2x2/block).
// Main K loop is unguarded (K rounded down to 32); guarded tail handles K=16.
// ---------------------------------------------------------------------------
struct GemmP {
  const float* A; const float* B; const float* bias; const float* res;
  float* C;
  int M, N, K;
  int lda, ldb, ldc;
  long sA1, sB1, sC1;
  long sA2, sB2, sC2;
  int bz2;
  int act;
};

// A 16x32 f16 fragment: elem j -> K = k0 + (j/8)*16 + half*8 + (j%8)
template<bool TA>
__device__ __forceinline__ v16h load_a_frag(const float* A, int lda, int row,
                                            int half, int k0) {
  v16h a;
  if (!TA) {
    // contiguous: j0..7 = base[0..7], j8..15 = base[16..23]; 16B aligned
    const float* base = A + (long)row * lda + k0 + (half << 3);
    const float4 f0 = *(const float4*)(base + 0);
    const float4 f1 = *(const float4*)(base + 4);
    const float4 f2 = *(const float4*)(base + 16);
    const float4 f3 = *(const float4*)(base + 20);
    a[0]  = (_Float16)f0.x; a[1]  = (_Float16)f0.y;
    a[2]  = (_Float16)f0.z; a[3]  = (_Float16)f0.w;
    a[4]  = (_Float16)f1.x; a[5]  = (_Float16)f1.y;
    a[6]  = (_Float16)f1.z; a[7]  = (_Float16)f1.w;
    a[8]  = (_Float16)f2.x; a[9]  = (_Float16)f2.y;
    a[10] = (_Float16)f2.z; a[11] = (_Float16)f2.w;
    a[12] = (_Float16)f3.x; a[13] = (_Float16)f3.y;
    a[14] = (_Float16)f3.z; a[15] = (_Float16)f3.w;
  } else {
    float t[16];
#pragma unroll
    for (int j = 0; j < 16; ++j) {
      const int kk = k0 + ((j >> 3) << 4) + (half << 3) + (j & 7);
      t[j] = A[(long)kk * lda + row];
    }
#pragma unroll
    for (int j = 0; j < 16; ++j) a[j] = (_Float16)t[j];
  }
  return a;
}

// B 32x16 f16 fragment: elem j -> K = k0 + (j/8)*16 + half*8 + (j%8)
template<bool TB>
__device__ __forceinline__ v16h load_b_frag(const float* B, int ldb, int col,
                                            int half, int k0) {
  v16h b;
  if (TB) {
    const float* base = B + (long)col * ldb + k0 + (half << 3);
    const float4 f0 = *(const float4*)(base + 0);
    const float4 f1 = *(const float4*)(base + 4);
    const float4 f2 = *(const float4*)(base + 16);
    const float4 f3 = *(const float4*)(base + 20);
    b[0]  = (_Float16)f0.x; b[1]  = (_Float16)f0.y;
    b[2]  = (_Float16)f0.z; b[3]  = (_Float16)f0.w;
    b[4]  = (_Float16)f1.x; b[5]  = (_Float16)f1.y;
    b[6]  = (_Float16)f1.z; b[7]  = (_Float16)f1.w;
    b[8]  = (_Float16)f2.x; b[9]  = (_Float16)f2.y;
    b[10] = (_Float16)f2.z; b[11] = (_Float16)f2.w;
    b[12] = (_Float16)f3.x; b[13] = (_Float16)f3.y;
    b[14] = (_Float16)f3.z; b[15] = (_Float16)f3.w;
  } else {
    float t[16];
#pragma unroll
    for (int j = 0; j < 16; ++j)
      t[j] = B[(long)(k0 + ((j >> 3) << 4) + (half << 3) + (j & 7)) * ldb + col];
#pragma unroll
    for (int j = 0; j < 16; ++j) b[j] = (_Float16)t[j];
  }
  return b;
}

template<bool TA, bool TB>
__global__ void __launch_bounds__(128) gemm_wmma_kernel(GemmP p) {
  const int wave = threadIdx.x >> 5;
  const int lane = threadIdx.x & 31;
  const int half = lane >> 4;
  const int l16  = lane & 15;

  const int z1 = blockIdx.z / p.bz2;
  const int z2 = blockIdx.z % p.bz2;
  const float* A  = p.A + (long)z1 * p.sA1 + (long)z2 * p.sA2;
  const float* Bp = p.B + (long)z1 * p.sB1 + (long)z2 * p.sB2;
  float*       C  = p.C + (long)z1 * p.sC1 + (long)z2 * p.sC2;
  const float* R  = p.res ? (p.res + (long)z1 * p.sC1 + (long)z2 * p.sC2) : nullptr;

  const int tm = blockIdx.y * 2 + (wave >> 1);
  const int tn = blockIdx.x * 2 + (wave & 1);
  if (tm * 16 >= p.M || tn * 16 >= p.N) return;

  const int rowA = tm * 16 + l16;   // A row (ISA: A row = lane%16)
  const int colB = tn * 16 + l16;   // B/C col (ISA: N = lane%16)

  const int kmain = p.K & ~31;
  v8f acc = {};
  for (int k0 = 0; k0 < kmain; k0 += 32) {
    if (k0 + 32 < kmain) {   // gfx1250 global_prefetch_b8 for next K slab
      __builtin_prefetch(TA ? &A[(long)(k0 + 32) * p.lda + rowA]
                            : &A[(long)rowA * p.lda + k0 + 32], 0, 1);
      __builtin_prefetch(TB ? &Bp[(long)colB * p.ldb + k0 + 32]
                            : &Bp[(long)(k0 + 32) * p.ldb + colB], 0, 1);
    }
    const v16h a = load_a_frag<TA>(A, p.lda, rowA, half, k0);
    const v16h b = load_b_frag<TB>(Bp, p.ldb, colB, half, k0);
    acc = __builtin_amdgcn_wmma_f32_16x16x32_f16(
        false, a, false, b, (short)0, acc, false, false);
  }
  if (kmain < p.K) {  // guarded tail (K % 32 != 0; e.g. K=16 dt projection)
    v16h a, b;
#pragma unroll
    for (int j = 0; j < 16; ++j) {
      const int kk = kmain + ((j >> 3) << 4) + (half << 3) + (j & 7);
      float v = 0.f;
      if (kk < p.K) v = TA ? A[(long)kk * p.lda + rowA] : A[(long)rowA * p.lda + kk];
      a[j] = (_Float16)v;
    }
#pragma unroll
    for (int j = 0; j < 16; ++j) {
      const int kk = kmain + ((j >> 3) << 4) + (half << 3) + (j & 7);
      float v = 0.f;
      if (kk < p.K) v = TB ? Bp[(long)colB * p.ldb + kk] : Bp[(long)kk * p.ldb + colB];
      b[j] = (_Float16)v;
    }
    acc = __builtin_amdgcn_wmma_f32_16x16x32_f16(
        false, a, false, b, (short)0, acc, false, false);
  }

  // D layout: VGPR r -> row = tm*16 + half*8 + r, col = colB
#pragma unroll
  for (int r = 0; r < 8; ++r) {
    const int m = tm * 16 + half * 8 + r;
    float v = acc[r];
    if (p.bias) v += p.bias[colB];
    v = act_apply(v, p.act);
    const long ci = (long)m * p.ldc + colB;
    if (R) v += R[ci];
    C[ci] = v;
  }
}

// ---------------------------------------------------------------------------
// Elementwise / reduction kernels
// ---------------------------------------------------------------------------

// Per-channel mean/var with arbitrary strides; block 256 per channel.
__global__ void __launch_bounds__(256) bn_stats_kernel(
    const float* x, float* stats, int C, int L, int NB,
    long sB, long sC, long sL) {
  const int c = blockIdx.x;
  const int t = threadIdx.x;
  const int cnt = NB * L;
  float s = 0.f, s2 = 0.f;
  for (int i = t; i < cnt; i += 256) {
    const int b = i / L, l = i - b * L;
    const float v = x[(long)b * sB + (long)c * sC + (long)l * sL];
    s += v; s2 += v * v;
  }
  __shared__ float r1[256], r2[256];
  r1[t] = s; r2[t] = s2; __syncthreads();
  for (int k = 128; k > 0; k >>= 1) {
    if (t < k) { r1[t] += r1[t + k]; r2[t] += r2[t + k]; }
    __syncthreads();
  }
  if (t == 0) {
    const float m = r1[0] / (float)cnt;
    stats[c] = m;
    stats[C + c] = r2[0] / (float)cnt - m * m;
  }
}

// x,y in [B,C,L] layout
__global__ void bn_apply_relu_cm(const float* x, const float* stats,
                                 const float* g, const float* b,
                                 float* y, int C, int L, long n) {
  const long i = (long)blockIdx.x * 256 + threadIdx.x;
  if (i >= n) return;
  const int c = (int)((i / L) % C);
  const float t = (x[i] - stats[c]) * rsqrtf(stats[C + c] + 1e-5f) * g[c] + b[c];
  y[i] = fmaxf(t, 0.f);
}

// x in [B,L,C] (channel-last), y written [B,C,L] (transposed) with BN+relu
__global__ void bn_apply_relu_cl2cm(const float* x, const float* stats,
                                    const float* g, const float* b,
                                    float* y, int C, int L, long n) {
  const long i = (long)blockIdx.x * 256 + threadIdx.x;
  if (i >= n) return;
  const long bb = i / ((long)C * L);
  const long rem = i - bb * (long)C * L;
  const int c = (int)(rem / L);
  const int l = (int)(rem - (long)c * L);
  const float xv = x[bb * (long)L * C + (long)l * C + c];
  const float t = (xv - stats[c]) * rsqrtf(stats[C + c] + 1e-5f) * g[c] + b[c];
  y[i] = fmaxf(t, 0.f);
}

// depthwise 3x3, pad 1, on [8,256,32,32]
__global__ void dwconv3x3_kernel(const float* x, const float* w, float* y, long n) {
  const long i = (long)blockIdx.x * 256 + threadIdx.x;
  if (i >= n) return;
  const int xx = (int)(i & 31);
  const int yy = (int)((i >> 5) & 31);
  const int c  = (int)((i >> 10) & 255);
  const long b = i >> 18;
  float acc = 0.f;
#pragma unroll
  for (int dy = 0; dy < 3; ++dy)
#pragma unroll
    for (int dx = 0; dx < 3; ++dx) {
      const int sy = yy + dy - 1, sx = xx + dx - 1;
      if (sy >= 0 && sy < 32 && sx >= 0 && sx < 32)
        acc += x[((b * 256 + c) * 32 + sy) * 32 + sx] * w[c * 9 + dy * 3 + dx];
    }
  y[i] = acc;
}

__global__ void add_pos_kernel(float* x, const float* pos, long n, long per) {
  const long i = (long)blockIdx.x * 256 + threadIdx.x;
  if (i >= n) return;
  x[i] += pos[i % per];
}

// RoPE on first 128 dims of [8,1024,256]; one thread per (b,l,pair)
__global__ void rope_kernel(float* x, long n) {
  const long i = (long)blockIdx.x * 256 + threadIdx.x;
  if (i >= n) return;
  const int pr = (int)(i & 63);
  const long bl = i >> 6;                // b*1024 + l
  const int l = (int)(bl & 1023);
  const float inv = __powf(10000.f, -(float)(2 * pr) / 128.f);
  const float fr = (float)l * inv;
  const float c = __cosf(fr), s = __sinf(fr);
  float* p = x + bl * 256 + 2 * pr;
  const float t0 = p[0], t1 = p[1];
  p[0] = t0 * c - t1 * s;
  p[1] = t1 * c + t0 * s;
}

// LayerNorm over last dim 256; block 256 per row
__global__ void __launch_bounds__(256) layernorm_kernel(
    const float* x, const float* g, const float* b, float* y) {
  const long row = blockIdx.x;
  const int t = threadIdx.x;
  const float v = x[row * 256 + t];
  __shared__ float red[256];
  red[t] = v; __syncthreads();
  for (int k = 128; k > 0; k >>= 1) { if (t < k) red[t] += red[t + k]; __syncthreads(); }
  const float m = red[0] / 256.f; __syncthreads();
  red[t] = (v - m) * (v - m); __syncthreads();
  for (int k = 128; k > 0; k >>= 1) { if (t < k) red[t] += red[t + k]; __syncthreads(); }
  const float var = red[0] / 256.f; __syncthreads();
  y[row * 256 + t] = (v - m) * rsqrtf(var + 1e-5f) * g[t] + b[t];
}

// softmax(x*scale) over rows of length L, in place
__global__ void __launch_bounds__(256) softmax_kernel(float* x, int L, float scale) {
  float* p = x + (long)blockIdx.x * L;
  const int t = threadIdx.x;
  __shared__ float red[256];
  float mx = -3.4e38f;
  for (int i = t; i < L; i += 256) mx = fmaxf(mx, p[i] * scale);
  red[t] = mx; __syncthreads();
  for (int k = 128; k > 0; k >>= 1) { if (t < k) red[t] = fmaxf(red[t], red[t + k]); __syncthreads(); }
  const float M = red[0]; __syncthreads();
  float s = 0.f;
  for (int i = t; i < L; i += 256) { const float e = __expf(p[i] * scale - M); p[i] = e; s += e; }
  red[t] = s; __syncthreads();
  for (int k = 128; k > 0; k >>= 1) { if (t < k) red[t] += red[t + k]; __syncthreads(); }
  const float inv = 1.f / red[0]; __syncthreads();
  for (int i = t; i < L; i += 256) p[i] *= inv;
}

// causal depthwise conv1d (K=4) + silu; xz [8,1024,1024] cols 0..511 are xs
__global__ void conv1d_silu_kernel(const float* xz, const float* w, const float* bias,
                                   float* out, long n) {
  const long i = (long)blockIdx.x * 256 + threadIdx.x;
  if (i >= n) return;
  const int d = (int)(i & 511);
  const long bl = i >> 9;             // b*1024 + l
  const int l = (int)(bl & 1023);
  float acc = bias[d];
#pragma unroll
  for (int t = 0; t < 4; ++t) {
    const int ls = l - 3 + t;
    if (ls >= 0) acc += xz[(bl - l + ls) * 1024 + d] * w[d * 4 + t];
  }
  out[i] = silu_f(acc);
}

// selective scan: grid=B, block=512 (one thread per d), state in registers
__global__ void __launch_bounds__(512) mamba_scan_kernel(
    const float* delta, const float* xdbl, const float* xsc,
    const float* A_log, float* ys) {
  const int b = blockIdx.x;
  const int d = threadIdx.x;
  float A[16], h[16];
#pragma unroll
  for (int s = 0; s < 16; ++s) { A[s] = -__expf(A_log[d * 16 + s]); h[s] = 0.f; }
  __shared__ float sB[16], sC[16];
  for (int l = 0; l < 1024; ++l) {
    const long row = (long)b * 1024 + l;
    if (d < 16) sB[d] = xdbl[row * 48 + 16 + d];
    else if (d < 32) sC[d - 16] = xdbl[row * 48 + 32 + (d - 16)];
    __syncthreads();
    const float dl = delta[row * 512 + d];
    const float u  = xsc[row * 512 + d];
    float y = 0.f;
#pragma unroll
    for (int s = 0; s < 16; ++s) {
      h[s] = __expf(dl * A[s]) * h[s] + dl * sB[s] * u;
      y += h[s] * sC[s];
    }
    ys[row * 512 + d] = y;
    __syncthreads();
  }
}

// y = (ys + xsc*D) * silu(z);  z = xz[..., 512+d]
__global__ void ygate_kernel(const float* ys, const float* xsc, const float* xz,
                             const float* Dp, float* out, long n) {
  const long i = (long)blockIdx.x * 256 + threadIdx.x;
  if (i >= n) return;
  const int d = (int)(i & 511);
  const long row = i >> 9;
  const float z = xz[row * 1024 + 512 + d];
  out[i] = (ys[i] + xsc[i] * Dp[d]) * silu_f(z);
}

__global__ void flip_l_kernel(const float* x, float* y, int L, int C, long n) {
  const long i = (long)blockIdx.x * 256 + threadIdx.x;
  if (i >= n) return;
  const long b = i / ((long)L * C);
  const long rem = i - b * (long)L * C;
  const int l = (int)(rem / C);
  const int c = (int)(rem - (long)l * C);
  y[i] = x[(b * L + (L - 1 - l)) * (long)C + c];
}

__global__ void flip_add_kernel(const float* x, float* y, int L, int C, long n) {
  const long i = (long)blockIdx.x * 256 + threadIdx.x;
  if (i >= n) return;
  const long b = i / ((long)L * C);
  const long rem = i - b * (long)L * C;
  const int l = (int)(rem / C);
  const int c = (int)(rem - (long)l * C);
  y[i] += x[(b * L + (L - 1 - l)) * (long)C + c];
}

// xs[b,l,c] = xt[b,l,c] + xmm[b,l,c] + cco[b,c,l] + cb[c]
__global__ void final_fuse_kernel(const float* xt, const float* xmm,
                                  const float* cco, const float* cb,
                                  float* xs, long n) {
  const long i = (long)blockIdx.x * 256 + threadIdx.x;
  if (i >= n) return;
  const int c = (int)(i & 255);
  const long bl = i >> 8;
  const int l = (int)(bl & 1023);
  const long b = bl >> 10;
  xs[i] = xt[i] + xmm[i] + cco[(b * 256 + c) * 1024 + l] + cb[c];
}

// mean over L, LayerNorm, classifier [256,16]; grid=B, block=256
__global__ void __launch_bounds__(256) pool_cls_kernel(
    const float* xs, const float* g, const float* bln,
    const float* W, const float* bias, float* out) {
  const int b = blockIdx.x;
  const int t = threadIdx.x;
  float s = 0.f;
  for (int l = 0; l < 1024; ++l) s += xs[((long)b * 1024 + l) * 256 + t];
  const float p = s * (1.f / 1024.f);
  __shared__ float red[256];
  __shared__ float nrm[256];
  red[t] = p; __syncthreads();
  for (int k = 128; k > 0; k >>= 1) { if (t < k) red[t] += red[t + k]; __syncthreads(); }
  const float m = red[0] / 256.f; __syncthreads();
  red[t] = (p - m) * (p - m); __syncthreads();
  for (int k = 128; k > 0; k >>= 1) { if (t < k) red[t] += red[t + k]; __syncthreads(); }
  const float var = red[0] / 256.f; __syncthreads();
  nrm[t] = (p - m) * rsqrtf(var + 1e-5f) * g[t] + bln[t];
  __syncthreads();
  if (t < 16) {
    float acc = bias[t];
    for (int c = 0; c < 256; ++c) acc += nrm[c] * W[c * 16 + t];
    out[b * 16 + t] = acc;
  }
}

// ---------------------------------------------------------------------------
// Host orchestration
// ---------------------------------------------------------------------------
static inline dim3 g1(long n) { return dim3((unsigned)((n + 255) / 256)); }

static void launch_gemm(hipStream_t stream,
    const float* A, const float* B, const float* bias, const float* res, float* C,
    int M, int N, int K, int lda, int ldb, int ldc,
    int nz1, long sA1, long sB1, long sC1,
    int bz2, long sA2, long sB2, long sC2,
    int transA, int transB, int act) {
  GemmP p;
  p.A = A; p.B = B; p.bias = bias; p.res = res; p.C = C;
  p.M = M; p.N = N; p.K = K; p.lda = lda; p.ldb = ldb; p.ldc = ldc;
  p.sA1 = sA1; p.sB1 = sB1; p.sC1 = sC1;
  p.sA2 = sA2; p.sB2 = sB2; p.sC2 = sC2;
  p.bz2 = bz2; p.act = act;
  dim3 grid((N + 31) / 32, (M + 31) / 32, nz1 * bz2);
  if (transA) {
    if (transB) gemm_wmma_kernel<true,  true ><<<grid, 128, 0, stream>>>(p);
    else        gemm_wmma_kernel<true,  false><<<grid, 128, 0, stream>>>(p);
  } else {
    if (transB) gemm_wmma_kernel<false, true ><<<grid, 128, 0, stream>>>(p);
    else        gemm_wmma_kernel<false, false><<<grid, 128, 0, stream>>>(p);
  }
}

extern "C" void kernel_launch(void* const* d_in, const int* in_sizes, int n_in,
                              void* d_out, int out_size, void* d_ws, size_t ws_size,
                              hipStream_t stream) {
  (void)in_sizes; (void)n_in; (void)out_size; (void)ws_size;
  auto P = [&](int i) { return (const float*)d_in[i]; };
  const float* x       = P(0);
  const float* conv1_w = P(1);
  const float* bn2_g   = P(2);  const float* bn2_b = P(3);
  const float* conv2_w = P(4);
  const float* bn3_g   = P(5);  const float* bn3_b = P(6);
  const float* conv3_w = P(7);
  const float* patch_w = P(8);  const float* patch_b = P(9);
  const float* pos     = P(10);
  const float* ln1_g   = P(11); const float* ln1_b = P(12);
  const float* Wqkv    = P(13);
  const float* Wo      = P(14); const float* bo = P(15);
  const float* ln2_g   = P(16); const float* ln2_b = P(17);
  const float* W1p     = P(18); const float* b1 = P(19);
  const float* W2p     = P(20); const float* b2 = P(21);
  // mamba params start at 22, mamba_back at 31 (9 each)
  const float* bnc_g   = P(40); const float* bnc_b = P(41);
  const float* convc_w = P(42); const float* convc_b = P(43);
  const float* cls_ln_g = P(44); const float* cls_ln_b = P(45);
  const float* cls_w   = P(46); const float* cls_b = P(47);

  float* Wd = (float*)d_ws;
  size_t off = 0;
  auto alloc = [&](size_t nf) { float* p = Wd + off; off += nf; return p; };
  float* x0     = alloc(2097152);   // [8,256,1024]  (also reused as xs_final)
  float* t1     = alloc(2097152);
  float* t2     = alloc(2097152);
  float* x4r    = alloc(524288);    // [8,64,1024]
  float* x3     = alloc(2097152);   // [8,1024,256]
  float* h1     = alloc(2097152);
  float* qkv    = alloc(12582912);  // [8,1024,1536]
  float* scores = alloc(33554432);  // [8,4,1024,1024]; reused as mamba arena
  float* o_     = alloc(4194304);   // [8,1024,512]
  float* xt     = alloc(2097152);
  float* h2     = alloc(2097152);
  float* mid    = alloc(1048576);   // [8192,128]
  float* xt2    = alloc(2097152);
  float* xmm    = alloc(2097152);
  float* xcT    = alloc(2097152);   // [8,256,1024]
  float* cco    = alloc(2097152);   // [8,256,1024]
  float* stats  = alloc(512);
  // mamba scratch aliases the (dead after attention) scores buffer
  float* mflip  = scores;             // [8,1024,256]
  float* mxz    = scores + 2097152;   // [8192,1024]
  float* mxsc   = scores + 10485760;  // [8192,512]
  float* mxdbl  = scores + 14680064;  // [8192,48]
  float* mdelta = scores + 15204352;  // [8192,512]
  float* mys    = scores + 19398656;  // [8192,512]
  float* myg    = scores + 23592960;  // [8192,512]
  float* mtmp   = scores + 27787264;  // [8,1024,256]

  // ---- conv stem ----
  // x0[b,e,hw] = conv1_w[e,c] * x[b,c,hw]
  launch_gemm(stream, conv1_w, x, nullptr, nullptr, x0,
              256, 1024, 64, 64, 1024, 1024,
              8, 0, 65536, 262144, 1, 0, 0, 0, 0, 0, ACT_NONE);
  bn_stats_kernel<<<256, 256, 0, stream>>>(x0, stats, 256, 1024, 8, 262144, 1024, 1);
  bn_apply_relu_cm<<<g1(2097152), 256, 0, stream>>>(x0, stats, bn2_g, bn2_b, t1, 256, 1024, 2097152);
  dwconv3x3_kernel<<<g1(2097152), 256, 0, stream>>>(t1, conv2_w, x0, 2097152);
  bn_stats_kernel<<<256, 256, 0, stream>>>(x0, stats, 256, 1024, 8, 262144, 1024, 1);
  bn_apply_relu_cm<<<g1(2097152), 256, 0, stream>>>(x0, stats, bn3_g, bn3_b, t2, 256, 1024, 2097152);
  // x4r[b,c,hw] = conv3_w[c,e]*t2[b,e,hw] + x
  launch_gemm(stream, conv3_w, t2, nullptr, x, x4r,
              64, 1024, 256, 256, 1024, 1024,
              8, 0, 262144, 65536, 1, 0, 0, 0, 0, 0, ACT_NONE);
  // tok[b,hw,e] = x4r[b,:,hw]^T @ patch_w + patch_b
  launch_gemm(stream, x4r, patch_w, patch_b, nullptr, x3,
              1024, 256, 64, 1024, 256, 256,
              8, 65536, 0, 262144, 1, 0, 0, 0, 1, 0, ACT_NONE);
  add_pos_kernel<<<g1(2097152), 256, 0, stream>>>(x3, pos, 2097152, 262144);
  rope_kernel<<<g1(524288), 256, 0, stream>>>(x3, 524288);

  // ---- transformer ----
  layernorm_kernel<<<8192, 256, 0, stream>>>(x3, ln1_g, ln1_b, h1);
  launch_gemm(stream, h1, Wqkv, nullptr, nullptr, qkv,
              8192, 1536, 256, 256, 1536, 1536,
              1, 0, 0, 0, 1, 0, 0, 0, 0, 0, ACT_NONE);
  // scores[b,h,q,k] = q . k  (batched over b (z1) and h (z2))
  launch_gemm(stream, qkv, qkv + 512, nullptr, nullptr, scores,
              1024, 1024, 128, 1536, 1536, 1024,
              8, 1572864, 1572864, 4194304,
              4, 128, 128, 1048576, 0, 1, ACT_NONE);
  softmax_kernel<<<32768, 256, 0, stream>>>(scores, 1024, 0.08838834764831845f);
  // o[b,q,h,d] = attn @ v
  launch_gemm(stream, scores, qkv + 1024, nullptr, nullptr, o_,
              1024, 128, 1024, 1024, 1536, 512,
              8, 4194304, 1572864, 524288,
              4, 1048576, 128, 128, 0, 0, ACT_NONE);
  launch_gemm(stream, o_, Wo, bo, x3, xt,
              8192, 256, 512, 512, 256, 256,
              1, 0, 0, 0, 1, 0, 0, 0, 0, 0, ACT_NONE);
  layernorm_kernel<<<8192, 256, 0, stream>>>(xt, ln2_g, ln2_b, h2);
  launch_gemm(stream, h2, W1p, b1, nullptr, mid,
              8192, 128, 256, 256, 128, 128,
              1, 0, 0, 0, 1, 0, 0, 0, 0, 0, ACT_GELU);
  launch_gemm(stream, mid, W2p, b2, xt, xt2,
              8192, 256, 128, 128, 256, 256,
              1, 0, 0, 0, 1, 0, 0, 0, 0, 0, ACT_NONE);

  // ---- bidirectional mamba ----
  auto run_mamba = [&](int pb, const float* xin, float* outC) {
    const float* in_proj  = P(pb + 0);
    const float* conv_w   = P(pb + 1);
    const float* conv_b   = P(pb + 2);
    const float* x_proj   = P(pb + 3);
    const float* dt_w     = P(pb + 4);
    const float* dt_b     = P(pb + 5);
    const float* A_log    = P(pb + 6);
    const float* Dp       = P(pb + 7);
    const float* out_proj = P(pb + 8);
    launch_gemm(stream, xin, in_proj, nullptr, nullptr, mxz,
                8192, 1024, 256, 256, 1024, 1024,
                1, 0, 0, 0, 1, 0, 0, 0, 0, 0, ACT_NONE);
    conv1d_silu_kernel<<<g1(4194304), 256, 0, stream>>>(mxz, conv_w, conv_b, mxsc, 4194304);
    launch_gemm(stream, mxsc, x_proj, nullptr, nullptr, mxdbl,
                8192, 48, 512, 512, 48, 48,
                1, 0, 0, 0, 1, 0, 0, 0, 0, 0, ACT_NONE);
    launch_gemm(stream, mxdbl, dt_w, dt_b, nullptr, mdelta,
                8192, 512, 16, 48, 512, 512,
                1, 0, 0, 0, 1, 0, 0, 0, 0, 0, ACT_SOFTPLUS);
    mamba_scan_kernel<<<8, 512, 0, stream>>>(mdelta, mxdbl, mxsc, A_log, mys);
    ygate_kernel<<<g1(4194304), 256, 0, stream>>>(mys, mxsc, mxz, Dp, myg, 4194304);
    launch_gemm(stream, myg, out_proj, nullptr, nullptr, outC,
                8192, 256, 512, 512, 256, 256,
                1, 0, 0, 0, 1, 0, 0, 0, 0, 0, ACT_NONE);
  };
  run_mamba(22, x3, xmm);                                         // forward
  flip_l_kernel<<<g1(2097152), 256, 0, stream>>>(x3, mflip, 1024, 256, 2097152);
  run_mamba(31, mflip, mtmp);                                     // backward
  flip_add_kernel<<<g1(2097152), 256, 0, stream>>>(mtmp, xmm, 1024, 256, 2097152);

  // ---- channel mixing branch ----
  bn_stats_kernel<<<256, 256, 0, stream>>>(xmm, stats, 256, 1024, 8, 262144, 1, 256);
  bn_apply_relu_cl2cm<<<g1(2097152), 256, 0, stream>>>(xmm, stats, bnc_g, bnc_b, xcT, 256, 1024, 2097152);
  launch_gemm(stream, convc_w, xcT, nullptr, nullptr, cco,
              256, 1024, 256, 256, 1024, 1024,
              8, 0, 262144, 262144, 1, 0, 0, 0, 0, 0, ACT_NONE);
  final_fuse_kernel<<<g1(2097152), 256, 0, stream>>>(xt2, xmm, cco, convc_b, x0, 2097152);

  // ---- head ----
  pool_cls_kernel<<<8, 256, 0, stream>>>(x0, cls_ln_g, cls_ln_b, cls_w, cls_b, (float*)d_out);
}
